// EdgeBiasedMHA_2207613190916
// MI455X (gfx1250) — matrix operand order repaired and checked by reading, not compiled, not run on hardware
//
#include <hip/hip_runtime.h>
#include <hip/hip_bf16.h>

#define D_MODEL 1024
#define HEADS   16
#define DK      64
#define BATCH   4
#define SEQ     1024
#define MROWS   (BATCH * SEQ)   // 4096

typedef __attribute__((ext_vector_type(16))) _Float16 v16h;
typedef __attribute__((ext_vector_type(8)))  _Float16 v8h;
typedef __attribute__((ext_vector_type(8)))  float    v8f;
typedef __attribute__((ext_vector_type(4)))  unsigned int u32x4;
typedef __attribute__((ext_vector_type(8)))  int          i32x8;
typedef __attribute__((ext_vector_type(4)))  int          i32x4;

__device__ __forceinline__ v8f wmma_f16(v16h a, v16h b, v8f c) {
  // D = A(16x32 f16) x B(32x16 f16) + C(16x16 f32)
  return __builtin_amdgcn_wmma_f32_16x16x32_f16(false, a, false, b, (short)0, c,
                                                false, false);
}

// A-fragment (16x32 f16, M x K) from row-major src (ld halfs per row).
// Lane L: row M = L&15; halves 0..7 -> K = c0..c0+7, halves 8..15 -> c0+16..c0+23
// with c0 = (L&16) ? 8 : 0   (ISA 16-bit A 16x32 layout).
__device__ __forceinline__ v16h load_a_frag(const _Float16* rowbase, int ld,
                                            int kbase, int lane) {
  const _Float16* p =
      rowbase + (size_t)(lane & 15) * ld + kbase + ((lane & 16) ? 8 : 0);
  union {
    v16h v;
    v8h h[2];
  } u;
  u.h[0] = *(const v8h*)(p);
  u.h[1] = *(const v8h*)(p + 16);
  return u.v;
}

// B-fragment (32x16 f16, K x N) from memory laid out [n][k] (k contiguous).
// Lane L: column N = L&15; halves 0..7 -> K = c0..c0+7, halves 8..15 -> c0+16..c0+23
// with c0 = kbase + ((L&16) ? 8 : 0).
__device__ __forceinline__ v16h load_b_frag(const _Float16* colbase, int ld,
                                            int kbase, int lane) {
  const _Float16* p =
      colbase + (size_t)(lane & 15) * ld + kbase + ((lane & 16) ? 8 : 0);
  union {
    v16h v;
    v8h h[2];
  } u;
  u.h[0] = *(const v8h*)(p);
  u.h[1] = *(const v8h*)(p + 16);
  return u.v;
}

// ---- Tensor Data Mover: 2D tile load Global -> LDS (D# per ISA 08 §8.3/8.4)
// data_size = 2 bytes (f16). dims/strides in element units.
// 6-arg builtin form: (g0 u32x4, g1 i32x8, g2 i32x4, g3 i32x4, i32x8, cpol)
__device__ __forceinline__ void tdm_load_2d(unsigned lds_addr, const void* gaddr,
                                            unsigned tensor_d0, unsigned tensor_d1,
                                            unsigned stride0,
                                            unsigned tile_d0, unsigned tile_d1) {
  unsigned long long ga = (unsigned long long)(uintptr_t)gaddr;
  u32x4 g0;
  g0[0] = 1u;                                        // count=1, user D#
  g0[1] = lds_addr;                                  // LDS byte address
  g0[2] = (unsigned)ga;                              // global_addr[31:0]
  g0[3] = ((unsigned)(ga >> 32) & 0x1FFFFFFu) |      // global_addr[56:32]
          (2u << 30);                                // type = 2 ("image")
  i32x8 g1;
  g1[0] = (int)(1u << 16);                           // data_size=1 -> 2 bytes
  g1[1] = (int)((tensor_d0 & 0xFFFFu) << 16);        // tensor_dim0[15:0]
  g1[2] = (int)((tensor_d0 >> 16) |                  // tensor_dim0[31:16]
                ((tensor_d1 & 0xFFFFu) << 16));      // tensor_dim1[15:0]
  g1[3] = (int)((tensor_d1 >> 16) |                  // tensor_dim1[31:16]
                ((tile_d0 & 0xFFFFu) << 16));        // tile_dim0
  g1[4] = (int)(tile_d1 & 0xFFFFu);                  // tile_dim1 (tile_dim2=0)
  g1[5] = (int)stride0;                              // tensor_dim0_stride lo
  g1[6] = 0;                                         // stride0 hi / stride1 lo
  g1[7] = 0;                                         // stride1 hi
  i32x4 z4 = {0, 0, 0, 0};                           // groups 2/3 unused (2D)
  i32x8 z8 = {0, 0, 0, 0, 0, 0, 0, 0};
  __builtin_amdgcn_tensor_load_to_lds(g0, g1, z4, z4, z8, 0);
}

// ---------------------------------------------------------------- conversions
__global__ void __launch_bounds__(256)
cvt_f32_to_f16(const float* __restrict__ src, _Float16* __restrict__ dst, int n) {
  int i = blockIdx.x * blockDim.x + threadIdx.x;
  int stride = gridDim.x * blockDim.x;
  for (; i < n; i += stride) dst[i] = (_Float16)src[i];
}

// W[k][c] (1024x1024 f32) -> Wt[c][k] f16 (so B-fragments are contiguous in k)
__global__ void __launch_bounds__(256)
cvt_transpose(const float* __restrict__ src, _Float16* __restrict__ dst) {
  int idx = blockIdx.x * blockDim.x + threadIdx.x;  // 1M threads
  int k = idx >> 10;
  int c = idx & 1023;
  dst[(size_t)c * D_MODEL + k] = (_Float16)src[idx];
}

// ------------------------------------------------------------------- GEMM
// C[M=4096, 1024] = A(f16 row-major) x Wt(f16, [n][k]) + bias
// One wave -> 16x64 output tile (4 C fragments), K loop step 32.
// MODE 0: f32 row-major out (+bias)            -> final projection
// MODE 1: f16 out at [b,h,n,dk]  (+bias)       -> Q, K
// MODE 2: f16 out at [b,h,dk,n]  (+bias)       -> V transposed
template <int MODE>
__global__ void __launch_bounds__(256)
gemm16(const _Float16* __restrict__ A, const _Float16* __restrict__ Wt,
       const float* __restrict__ bias, void* __restrict__ out) {
  int wave = (blockIdx.x * blockDim.x + threadIdx.x) >> 5;
  int lane = threadIdx.x & 31;
  const int NT = D_MODEL / 64;  // 16 col tiles
  int mt = wave / NT;
  int nt = wave % NT;
  if (mt >= MROWS / 16) return;

  const _Float16* Arow = A + (size_t)mt * 16 * D_MODEL;

  v8f acc[4];
#pragma unroll
  for (int f = 0; f < 4; ++f)
#pragma unroll
    for (int r = 0; r < 8; ++r) acc[f][r] = 0.f;

  for (int k = 0; k < D_MODEL; k += 32) {
    v16h a = load_a_frag(Arow, D_MODEL, k, lane);
#pragma unroll
    for (int f = 0; f < 4; ++f) {
      v16h b = load_b_frag(Wt + (size_t)(nt * 64 + f * 16) * D_MODEL, D_MODEL,
                           k, lane);
      acc[f] = wmma_f16(a, b, acc[f]);
    }
  }

  int col = lane & 15;
  int roff = (lane & 16) ? 8 : 0;
#pragma unroll
  for (int f = 0; f < 4; ++f) {
#pragma unroll
    for (int r = 0; r < 8; ++r) {
      int m = mt * 16 + r + roff;            // global row (b*1024 + n)
      int c = nt * 64 + f * 16 + col;        // global col (h*64 + d)
      float v = acc[f][r] + bias[c];
      if (MODE == 0) {
        ((float*)out)[(size_t)m * D_MODEL + c] = v;
      } else {
        int b = m >> 10, n = m & 1023;
        int h = c >> 6, d = c & 63;
        if (MODE == 1)
          ((_Float16*)out)[(((size_t)(b * HEADS + h)) * SEQ + n) * DK + d] =
              (_Float16)v;
        else
          ((_Float16*)out)[(((size_t)(b * HEADS + h)) * DK + d) * SEQ + n] =
              (_Float16)v;
      }
    }
  }
}

// --------------------------------------------------------------- attention
// 8 waves per block, all on the same (b,h); wave w owns one 16-query tile.
// 32-key blocks: K/V tiles staged Global->LDS by the Tensor Data Mover
// (double buffered, wave 0 issues, s_wait_tensorcnt + barriers synchronize),
// flash-style online softmax, P round-trips through per-wave LDS.
__global__ void __launch_bounds__(256)
attn_kernel(const _Float16* __restrict__ Qh, const _Float16* __restrict__ Kh,
            const _Float16* __restrict__ Vth, const float* __restrict__ Bij,
            const unsigned char* __restrict__ mask,
            _Float16* __restrict__ Ctx) {
  int wave = (blockIdx.x * blockDim.x + threadIdx.x) >> 5;
  int warp = threadIdx.x >> 5;
  int lane = threadIdx.x & 31;
  int qt = wave & 63;   // SEQ/16 query tiles
  int bh = wave >> 6;   // b*HEADS + h
  int b = bh >> 4;
  int h = bh & 15;

  const _Float16* Qp = Qh + ((size_t)bh * SEQ + qt * 16) * DK;
  const _Float16* Kp = Kh + (size_t)bh * SEQ * DK;
  const _Float16* Vp = Vth + (size_t)bh * DK * SEQ;
  const float* Bp = Bij + ((size_t)bh * SEQ + qt * 16) * SEQ;
  const unsigned char* mp = mask + (size_t)b * SEQ;

  __shared__ __align__(128) _Float16 Ktile[2][32 * DK];  // [key][d], 4KB each
  __shared__ __align__(128) _Float16 Vtile[2][DK * 32];  // [d][key], 4KB each
  __shared__ __align__(32) _Float16 Pbuf[8][16 * 32];
  _Float16* pb = Pbuf[warp];

  unsigned kldsbase = (unsigned)(uintptr_t)(void*)&Ktile[0][0];
  unsigned vldsbase = (unsigned)(uintptr_t)(void*)&Vtile[0][0];

  int col = lane & 15;
  int roff = (lane & 16) ? 8 : 0;

  // Q fragments (reused over all key blocks)
  v16h aq0 = load_a_frag(Qp, DK, 0, lane);
  v16h aq1 = load_a_frag(Qp, DK, 32, lane);

  v8f acc[4];
  float mrun[8], lrun[8];
#pragma unroll
  for (int r = 0; r < 8; ++r) { mrun[r] = -1e30f; lrun[r] = 0.f; }
#pragma unroll
  for (int f = 0; f < 4; ++f)
#pragma unroll
    for (int r = 0; r < 8; ++r) acc[f][r] = 0.f;

  const float scale = 0.125f;  // 1/sqrt(64)

  // prologue: TDM-load first K/V tile into buffer 0
  if (warp == 0) {
    // K: tensor [key][d] -> tile 32 keys x 64 d, LDS [32][64]
    tdm_load_2d(kldsbase, Kp, DK, SEQ, DK, DK, 32);
    // V: tensor [d][key] -> tile 64 d x 32 keys, LDS [64][32]
    tdm_load_2d(vldsbase, Vp, SEQ, DK, SEQ, 32, DK);
  }

  int cur = 0;
  for (int kb = 0; kb < SEQ; kb += 32) {
    if (warp == 0) __builtin_amdgcn_s_wait_tensorcnt((short)0);
    __syncthreads();  // buffer `cur` ready for all waves

    // kick off next tile into the other buffer while we compute
    if (warp == 0 && (kb + 32) < SEQ) {
      int nxt = cur ^ 1;
      tdm_load_2d(kldsbase + (unsigned)nxt * 32 * DK * 2,
                  (const char*)(const void*)Kp + (size_t)(kb + 32) * DK * 2,
                  DK, SEQ, DK, DK, 32);
      tdm_load_2d(vldsbase + (unsigned)nxt * DK * 32 * 2,
                  (const char*)(const void*)Vp + (size_t)(kb + 32) * 2,
                  SEQ, DK, SEQ, 32, DK);
    }

    const _Float16* Kl = Ktile[cur];  // [32 keys][64 d]
    const _Float16* Vl = Vtile[cur];  // [64 d][32 keys]

    if (kb + 32 < SEQ)
      __builtin_prefetch(Bp + (size_t)roff * SEQ + kb + 32, 0, 1);

    // ---- two 16x16 score tiles: S = Q K^T (contraction over dk=64)
    v8f s0, s1;
    {
      v16h b0 = load_b_frag(Kl, DK, 0, lane);
      v16h b1 = load_b_frag(Kl, DK, 32, lane);
      v8f c;
#pragma unroll
      for (int r = 0; r < 8; ++r) c[r] = 0.f;
      c = wmma_f16(aq0, b0, c);
      c = wmma_f16(aq1, b1, c);
      s0 = c;
    }
    {
      v16h b0 = load_b_frag(Kl + 16 * DK, DK, 0, lane);
      v16h b1 = load_b_frag(Kl + 16 * DK, DK, 32, lane);
      v8f c;
#pragma unroll
      for (int r = 0; r < 8; ++r) c[r] = 0.f;
      c = wmma_f16(aq0, b0, c);
      c = wmma_f16(aq1, b1, c);
      s1 = c;
    }

    // ---- scale + edge bias + mask (C layout: lane holds key col, 8 q rows)
    int key0 = kb + col;
    int key1 = kb + 16 + col;
    float ok0 = mp[key0] ? 0.f : 1.f;
    float ok1 = mp[key1] ? 0.f : 1.f;
#pragma unroll
    for (int r = 0; r < 8; ++r) {
      size_t row = (size_t)(r + roff) * SEQ;
      float v0 = s0[r] * scale + Bp[row + key0];
      float v1 = s1[r] * scale + Bp[row + key1];
      s0[r] = ok0 == 0.f ? v0 : -10000.f;
      s1[r] = ok1 == 0.f ? v1 : -10000.f;
    }

    // ---- online softmax: rows live across each 16-lane half
    float tm[8];
#pragma unroll
    for (int r = 0; r < 8; ++r) tm[r] = fmaxf(s0[r], s1[r]);
#pragma unroll
    for (int m = 1; m < 16; m <<= 1)
#pragma unroll
      for (int r = 0; r < 8; ++r)
        tm[r] = fmaxf(tm[r], __shfl_xor(tm[r], m, 16));

    float nscale[8];
#pragma unroll
    for (int r = 0; r < 8; ++r) {
      float nm = fmaxf(mrun[r], tm[r]);
      nscale[r] = __expf(mrun[r] - nm);
      mrun[r] = nm;
    }

    float ps[8];
#pragma unroll
    for (int r = 0; r < 8; ++r) {
      float p0 = __expf(s0[r] - mrun[r]);
      float p1 = __expf(s1[r] - mrun[r]);
      s0[r] = p0;
      s1[r] = p1;
      ps[r] = p0 + p1;
    }
#pragma unroll
    for (int m = 1; m < 16; m <<= 1)
#pragma unroll
      for (int r = 0; r < 8; ++r) ps[r] += __shfl_xor(ps[r], m, 16);
#pragma unroll
    for (int r = 0; r < 8; ++r) lrun[r] = lrun[r] * nscale[r] + ps[r];

#pragma unroll
    for (int f = 0; f < 4; ++f)
#pragma unroll
      for (int r = 0; r < 8; ++r) acc[f][r] *= nscale[r];

    // ---- P (C layout) -> LDS row-major [q][32 keys] -> A fragment
#pragma unroll
    for (int r = 0; r < 8; ++r) {
      pb[(r + roff) * 32 + col] = (_Float16)s0[r];
      pb[(r + roff) * 32 + 16 + col] = (_Float16)s1[r];
    }
    asm volatile("s_wait_dscnt 0" ::: "memory");
    v16h ap = load_a_frag(pb, 32, 0, lane);

    // ---- O += P x V  (V tile in LDS: [d][key], key contiguous)
#pragma unroll
    for (int f = 0; f < 4; ++f) {
      v16h bv = load_b_frag(Vl + (size_t)(f * 16) * 32, 32, 0, lane);
      acc[f] = wmma_f16(ap, bv, acc[f]);
    }

    __syncthreads();  // everyone done with buffer `cur` before TDM reuses it
    cur ^= 1;
  }

  // ---- epilogue: normalize, store f16 context at [b, n, h*64 + d]
#pragma unroll
  for (int f = 0; f < 4; ++f) {
#pragma unroll
    for (int r = 0; r < 8; ++r) {
      float inv = 1.f / lrun[r];
      int q = qt * 16 + r + roff;
      int d = f * 16 + col;
      Ctx[((size_t)b * SEQ + q) * D_MODEL + h * DK + d] =
          (_Float16)(acc[f][r] * inv);
    }
  }
}

// ------------------------------------------------------------------ launch
extern "C" void kernel_launch(void* const* d_in, const int* in_sizes, int n_in,
                              void* d_out, int out_size, void* d_ws,
                              size_t ws_size, hipStream_t stream) {
  const float* X = (const float*)d_in[0];
  const float* Bij = (const float*)d_in[1];
  const unsigned char* mask = (const unsigned char*)d_in[2];
  const float* Wq = (const float*)d_in[3];
  const float* bq = (const float*)d_in[4];
  const float* Wk = (const float*)d_in[5];
  const float* bk = (const float*)d_in[6];
  const float* Wv = (const float*)d_in[7];
  const float* bv = (const float*)d_in[8];
  const float* Wo = (const float*)d_in[9];
  const float* bo = (const float*)d_in[10];

  char* ws = (char*)d_ws;
  const size_t MB = 1u << 20;
  _Float16* Xh  = (_Float16*)(ws + 0 * MB);   // 8 MB
  _Float16* Wqt = (_Float16*)(ws + 8 * MB);   // 2 MB
  _Float16* Wkt = (_Float16*)(ws + 10 * MB);  // 2 MB
  _Float16* Wvt = (_Float16*)(ws + 12 * MB);  // 2 MB
  _Float16* Wot = (_Float16*)(ws + 14 * MB);  // 2 MB
  _Float16* Qh  = (_Float16*)(ws + 16 * MB);  // 8 MB
  _Float16* Kh  = (_Float16*)(ws + 24 * MB);  // 8 MB
  _Float16* Vth = (_Float16*)(ws + 32 * MB);  // 8 MB
  _Float16* Ctx = (_Float16*)(ws + 40 * MB);  // 8 MB

  // f32 -> f16 conversions
  cvt_f32_to_f16<<<4096, 256, 0, stream>>>(X, Xh, MROWS * D_MODEL);
  cvt_transpose<<<4096, 256, 0, stream>>>(Wq, Wqt);
  cvt_transpose<<<4096, 256, 0, stream>>>(Wk, Wkt);
  cvt_transpose<<<4096, 256, 0, stream>>>(Wv, Wvt);
  cvt_transpose<<<4096, 256, 0, stream>>>(Wo, Wot);

  // QKV projections (WMMA GEMM)
  gemm16<1><<<512, 256, 0, stream>>>(Xh, Wqt, bq, Qh);
  gemm16<1><<<512, 256, 0, stream>>>(Xh, Wkt, bk, Kh);
  gemm16<2><<<512, 256, 0, stream>>>(Xh, Wvt, bv, Vth);

  // flash attention with edge bias (TDM-staged K/V)
  attn_kernel<<<512, 256, 0, stream>>>(Qh, Kh, Vth, Bij, mask, Ctx);

  // output projection -> f32 d_out
  gemm16<0><<<512, 256, 0, stream>>>(Ctx, Wot, bo, d_out);
}
